// MHAAttention_9113920602381
// MI455X (gfx1250) — matrix operand order repaired and checked by reading, not compiled, not run on hardware
//
#include <hip/hip_runtime.h>

typedef unsigned short u16;
typedef __attribute__((ext_vector_type(16))) __bf16 bf16x16;
typedef __attribute__((ext_vector_type(8)))  float   f32x8;

#define B_     64
#define L_     200
#define HID_   256
#define NH_    8
#define HD_    2048
#define MQ_    12800   // B*L
#define MAREA_ 990
#define MPAD_  992

// ---------- bf16 helpers (RNE) ----------
__device__ __forceinline__ u16 f2bf(float x) {
  unsigned u = __float_as_uint(x);
  u += 0x7fffu + ((u >> 16) & 1u);
  return (u16)(u >> 16);
}
__device__ __forceinline__ float bf2f(u16 h) {
  return __uint_as_float(((unsigned)h) << 16);
}

union FragU { struct { uint4 lo, hi; } q; bf16x16 v; };

// A operand, 16x32 bf16: lane m = l&15; k = j + 8*(j>=8) + 8*half  (ISA 7.12.2)
__device__ __forceinline__ bf16x16 load_afrag(const u16* p, int half) {
  FragU f;
  f.q.lo = *reinterpret_cast<const uint4*>(p + 8 * half);
  f.q.hi = *reinterpret_cast<const uint4*>(p + 16 + 8 * half);
  return f.v;
}
// B operand, 32x16 bf16: lane n = l&15; k = j + 8*(j>=8) + 8*half (same as A)
__device__ __forceinline__ bf16x16 load_bfrag(const u16* p, int half) {
  FragU f;
  f.q.lo = *reinterpret_cast<const uint4*>(p + 8 * half);
  f.q.hi = *reinterpret_cast<const uint4*>(p + 16 + 8 * half);
  return f.v;
}
__device__ __forceinline__ bf16x16 zero_frag() {
  FragU f;
  f.q.lo = make_uint4(0, 0, 0, 0);
  f.q.hi = make_uint4(0, 0, 0, 0);
  return f.v;
}
__device__ __forceinline__ f32x8 wmma_bf16(bf16x16 a, bf16x16 b, f32x8 c) {
  return __builtin_amdgcn_wmma_f32_16x16x32_bf16(false, a, false, b, (short)0, c,
                                                 false, false);
}

// LDS byte offset of a generic pointer derived from __shared__ (low 32 bits of flat addr)
__device__ __forceinline__ unsigned lds_off(const void* p) {
  return (unsigned)(unsigned long long)p;
}
// gfx1250 async global->LDS copy, 16B per enabled lane (ISA 15.18.3 op 98)
__device__ __forceinline__ void async_copy_b128(unsigned lds, const void* gaddr) {
  asm volatile("global_load_async_to_lds_b128 %0, %1, off"
               :: "v"(lds), "v"(gaddr) : "memory");
}
__device__ __forceinline__ void wait_async0() {
  asm volatile("s_wait_asynccnt 0x0" ::: "memory");
}

// ---------- elementwise fp32 -> bf16 ----------
__global__ __launch_bounds__(256) void f2bf_kernel(const float* in, u16* out, int n) {
  int i = blockIdx.x * 256 + threadIdx.x;
  if (i < n) out[i] = f2bf(in[i]);
}

// ---------- weight swizzle: W[K,N] row-major -> [(K/32)][N][32] bf16 ----------
__global__ __launch_bounds__(256) void swizzleB_kernel(const float* W, u16* out,
                                                       int K, int N) {
  int i = blockIdx.x * 256 + threadIdx.x;
  if (i < K * N) {
    int k = i / N, n = i - k * N;
    out[((size_t)(k >> 5) * N + n) * 32 + (k & 31)] = f2bf(W[i]);
  }
}

// ---------- WMMA GEMM: C[M,N] = A_bf16[M,K] @ Bsw + bias; optional scale/relu ----------
// Block 128 thr (4 waves), tile 64x64. Ping-pong double buffering (ksteps always
// even): consume buffer 0 while loading buffer 1 and vice versa -> no register
// rotation moves, no WMMA->VALU WAR hazard NOPs, loads a full k-step ahead.
__global__ __launch_bounds__(128) void gemm_bf16_kernel(
    const u16* __restrict__ A, const u16* __restrict__ Bsw,
    const float* __restrict__ bias, float* __restrict__ Cf, u16* __restrict__ Cb,
    int M, int N, int K, float scale, int relu) {
  const int tid = threadIdx.x;
  const int w = tid >> 5, lane = tid & 31;
  const int l16 = lane & 15, half = lane >> 4;
  const int rbase = blockIdx.x * 64 + w * 16;
  const int cbase = blockIdx.y * 64;

  const u16* arow = A + (size_t)(rbase + l16) * K;
  const int ksteps = K >> 5;  // 8 / 32 / 64 here: always even

  f32x8 c[4] = {{0,0,0,0,0,0,0,0},{0,0,0,0,0,0,0,0},
                {0,0,0,0,0,0,0,0},{0,0,0,0,0,0,0,0}};
  bf16x16 a0, a1, b0[4], b1[4];
  a0 = load_afrag(arow, half);
#pragma unroll
  for (int nt = 0; nt < 4; ++nt)
    b0[nt] = load_bfrag(Bsw + (size_t)(cbase + nt * 16 + l16) * 32, half);

  for (int ks = 0; ks < ksteps; ks += 2) {
    // stage ks+1 into buffer 1
    a1 = load_afrag(arow + (ks + 1) * 32, half);
#pragma unroll
    for (int nt = 0; nt < 4; ++nt)
      b1[nt] = load_bfrag(Bsw + ((size_t)(ks + 1) * N + cbase + nt * 16 + l16) * 32,
                          half);
    if (ks + 3 < ksteps)  // pull B panels two steps out toward L2/L0
      __builtin_prefetch(Bsw + ((size_t)(ks + 3) * N + cbase + l16 * 4) * 32, 0, 1);
#pragma unroll
    for (int nt = 0; nt < 4; ++nt) c[nt] = wmma_bf16(a0, b0[nt], c[nt]);
    // stage ks+2 into buffer 0 (if any)
    if (ks + 2 < ksteps) {
      a0 = load_afrag(arow + (ks + 2) * 32, half);
#pragma unroll
      for (int nt = 0; nt < 4; ++nt)
        b0[nt] = load_bfrag(Bsw + ((size_t)(ks + 2) * N + cbase + nt * 16 + l16) * 32,
                            half);
    }
#pragma unroll
    for (int nt = 0; nt < 4; ++nt) c[nt] = wmma_bf16(a1, b1[nt], c[nt]);
  }
#pragma unroll
  for (int nt = 0; nt < 4; ++nt) {
    int n = cbase + nt * 16 + l16;
    float bi = bias[n];
#pragma unroll
    for (int r = 0; r < 8; ++r) {
      int row = rbase + r + 8 * half;
      float v = (c[nt][r] + bi) * scale;
      if (relu) v = fmaxf(v, 0.f);
      if (Cf) Cf[(size_t)row * N + n] = v;
      if (Cb) Cb[(size_t)row * N + n] = f2bf(v);
    }
  }
}

// ---------- area pooling: K max-pool, V sum-pool, widths 1..5 ----------
__global__ __launch_bounds__(256) void pool_kernel(const u16* __restrict__ Kp,
                                                   const u16* __restrict__ Vp,
                                                   u16* __restrict__ Ka,
                                                   u16* __restrict__ Va) {
  const int bh = blockIdx.x, d = threadIdx.x;
  const int b = bh >> 3, h = bh & 7;
  const int OFF[5] = {0, 200, 399, 597, 794};
  float kbuf[5], vbuf[5];
  for (int l = 0; l < L_; ++l) {
#pragma unroll
    for (int i = 4; i >= 1; --i) { kbuf[i] = kbuf[i - 1]; vbuf[i] = vbuf[i - 1]; }
    size_t src = (size_t)(b * L_ + l) * HD_ + h * HID_ + d;
    kbuf[0] = bf2f(Kp[src]);
    vbuf[0] = bf2f(Vp[src]);
    float kmax = kbuf[0], vsum = vbuf[0];
#pragma unroll
    for (int wdt = 1; wdt <= 5; ++wdt) {
      if (wdt > 1) { kmax = fmaxf(kmax, kbuf[wdt - 1]); vsum += vbuf[wdt - 1]; }
      if (l >= wdt - 1) {
        int m = OFF[wdt - 1] + (l - wdt + 1);
        size_t dst = ((size_t)bh * MPAD_ + m) * HID_ + d;
        Ka[dst] = f2bf(kmax);
        Va[dst] = f2bf(vsum);
      }
    }
  }
  size_t padb = ((size_t)bh * MPAD_ + MAREA_) * HID_ + d;
  Ka[padb] = 0; Ka[padb + HID_] = 0;
  Va[padb] = 0; Va[padb + HID_] = 0;
}

// ---------- fused flash area-attention ----------
// grid (4, 512): x = 64-row q block, y = b*H+h. Block 128 thr = 4 waves.
// Q fragments live in registers (loaded once); K tile async-copied to LDS and
// shared by all 4 waves; V tile transposed into LDS for contiguous B-fragments.
__global__ __launch_bounds__(128) void flash_kernel(const u16* __restrict__ Q,
                                                    const u16* __restrict__ Ka,
                                                    const u16* __restrict__ Va,
                                                    u16* __restrict__ ctx) {
  __shared__ __align__(16) u16 Ks[32 * 264];   // K tile, padded stride 264
  __shared__ __align__(16) u16 VsT[256 * 40];  // V tile transposed [d][m]
  __shared__ __align__(16) u16 Ps[4 * 16 * 40];// per-wave P tile

  const int tid = threadIdx.x;
  const int w = tid >> 5, lane = tid & 31;
  const int l16 = lane & 15, half = lane >> 4;
  const int bh = blockIdx.y, b = bh >> 3, h = bh & 7;
  const int qbase = blockIdx.x * 64;

  // Q fragments in registers: lane holds row (qbase + 16w + l16); guard L=200 tail.
  const int qr = qbase + w * 16 + l16;
  const bool qvalid = (qr < L_);
  bf16x16 qf[8];
  {
    const u16* qp = Q + (size_t)(b * L_ + (qvalid ? qr : 0)) * HD_ + h * HID_;
#pragma unroll
    for (int ds = 0; ds < 8; ++ds)
      qf[ds] = qvalid ? load_afrag(qp + ds * 32, half) : zero_frag();
  }

  f32x8 o[16];
#pragma unroll
  for (int dt = 0; dt < 16; ++dt) o[dt] = f32x8{0,0,0,0,0,0,0,0};
  float mrow[8], lsum[8];
#pragma unroll
  for (int r = 0; r < 8; ++r) { mrow[r] = -1e30f; lsum[r] = 0.f; }

  for (int mt = 0; mt < 31; ++mt) {
    __syncthreads();  // all waves done with Ks/VsT from previous tile

    // async-stage K tile cols 0..127 (32x128 bf16) into padded LDS: 4 x 16B chunks per thread
    const u16* kgbase = Ka + ((size_t)bh * MPAD_ + mt * 32) * HID_;
#pragma unroll
    for (int i = 0; i < 4; ++i) {
      int cch = i * 128 + tid;
      int row = cch >> 4, coff = (cch & 15) * 8;
      async_copy_b128(lds_off(&Ks[row * 264 + coff]), kgbase + row * HID_ + coff);
    }
    // V tile transpose into LDS via VGPRs (layout transform)
    const u16* vgbase = Va + ((size_t)bh * MPAD_ + mt * 32) * HID_;
    for (int i = 0; i < 64; ++i) {
      int e = i * 128 + tid;
      int ml = e >> 8, d = e & 255;
      VsT[d * 40 + ml] = vgbase[(size_t)ml * HID_ + d];
    }
    // prefetch next tile toward L2 while this one is consumed
    if (mt < 30) {
      const u16* kn = Ka + ((size_t)bh * MPAD_ + (mt + 1) * 32) * HID_;
      const u16* vn = Va + ((size_t)bh * MPAD_ + (mt + 1) * 32) * HID_;
      __builtin_prefetch(kn + tid * 64, 0, 1);
      __builtin_prefetch(vn + tid * 64, 0, 1);
    }
    wait_async0();
    __syncthreads();

    // logits [16q x 32m] from register Q frags and LDS K tile
    f32x8 c0 = {0,0,0,0,0,0,0,0}, c1 = {0,0,0,0,0,0,0,0};
#pragma unroll
    for (int ds = 0; ds < 8; ++ds) {
      bf16x16 b0 = load_bfrag(Ks + l16 * 264 + ds * 32, half);
      bf16x16 b1 = load_bfrag(Ks + (16 + l16) * 264 + ds * 32, half);
      c0 = wmma_bf16(qf[ds], b0, c0);
      c1 = wmma_bf16(qf[ds], b1, c1);
    }

    // online softmax; C layout: lane n=l16, reg r -> row r+8*half
    const int m0 = mt * 32 + l16, m1 = m0 + 16;
#pragma unroll
    for (int r = 0; r < 8; ++r) {
      float x0 = (m0 < MAREA_) ? c0[r] : -1e30f;
      float x1 = (m1 < MAREA_) ? c1[r] : -1e30f;
      float mx = fmaxf(x0, x1);
      mx = fmaxf(mx, __shfl_xor(mx, 1));
      mx = fmaxf(mx, __shfl_xor(mx, 2));
      mx = fmaxf(mx, __shfl_xor(mx, 4));
      mx = fmaxf(mx, __shfl_xor(mx, 8));
      float nm = fmaxf(mrow[r], mx);
      float corr = __expf(mrow[r] - nm);
      float p0 = __expf(x0 - nm), p1 = __expf(x1 - nm);
      float ps = p0 + p1;
      ps += __shfl_xor(ps, 1);
      ps += __shfl_xor(ps, 2);
      ps += __shfl_xor(ps, 4);
      ps += __shfl_xor(ps, 8);
      lsum[r] = lsum[r] * corr + ps;
      mrow[r] = nm;
#pragma unroll
      for (int dt = 0; dt < 16; ++dt) o[dt][r] *= corr;
      int prow = w * 640 + (r + 8 * half) * 40;
      Ps[prow + l16] = f2bf(p0);
      Ps[prow + 16 + l16] = f2bf(p1);
    }
    __syncthreads();  // P visible as A-fragment source

    // O += P(16x32) * V(32x256); VsT rows are B-frag contiguous in m
    bf16x16 ap = load_afrag(Ps + w * 640 + l16 * 40, half);
#pragma unroll
    for (int dt = 0; dt < 16; ++dt) {
      bf16x16 bv = load_bfrag(VsT + (dt * 16 + l16) * 40, half);
      o[dt] = wmma_bf16(ap, bv, o[dt]);
    }
  }

  // write ctx (bf16, feeds Wo GEMM); O rows: r+8*half, cols: dt*16+l16
#pragma unroll
  for (int r = 0; r < 8; ++r) {
    float inv = 1.f / lsum[r];
    int q = qbase + w * 16 + r + 8 * half;
    if (q < L_) {
      size_t basep = (size_t)(b * L_ + q) * HD_ + h * HID_;
#pragma unroll
      for (int dt = 0; dt < 16; ++dt)
        ctx[basep + dt * 16 + l16] = f2bf(o[dt][r] * inv);
    }
  }
}

// ---------- residual add + layernorm (eps 1e-5), dual fp32/bf16 output ----------
__global__ __launch_bounds__(256) void addln_kernel(const float* __restrict__ x,
                                                    const float* __restrict__ y,
                                                    float* __restrict__ of,
                                                    u16* __restrict__ ob) {
  const int row = blockIdx.x, t = threadIdx.x;
  const int lane = t & 31, wid = t >> 5;
  __shared__ float red[8];
  __shared__ float stat[2];
  float v = x[(size_t)row * HID_ + t] + y[(size_t)row * HID_ + t];
  float s = v;
#pragma unroll
  for (int m = 16; m >= 1; m >>= 1) s += __shfl_xor(s, m);
  if (lane == 0) red[wid] = s;
  __syncthreads();
  if (t == 0) {
    float tt = 0;
    for (int i = 0; i < 8; ++i) tt += red[i];
    stat[0] = tt * (1.f / 256.f);
  }
  __syncthreads();
  float mean = stat[0];
  float d = v - mean;
  s = d * d;
#pragma unroll
  for (int m = 16; m >= 1; m >>= 1) s += __shfl_xor(s, m);
  if (lane == 0) red[wid] = s;
  __syncthreads();
  if (t == 0) {
    float tt = 0;
    for (int i = 0; i < 8; ++i) tt += red[i];
    stat[1] = tt * (1.f / 256.f);
  }
  __syncthreads();
  float o = d * rsqrtf(stat[1] + 1e-5f);
  if (of) of[(size_t)row * HID_ + t] = o;
  if (ob) ob[(size_t)row * HID_ + t] = f2bf(o);
}

// ================= host launcher =================
extern "C" void kernel_launch(void* const* d_in, const int* in_sizes, int n_in,
                              void* d_out, int out_size, void* d_ws, size_t ws_size,
                              hipStream_t stream) {
  (void)in_sizes; (void)n_in; (void)out_size; (void)ws_size;
  const float* hidden = (const float*)d_in[1];
  const float* Wq = (const float*)d_in[2];  const float* bq = (const float*)d_in[3];
  const float* Wk = (const float*)d_in[4];  const float* bk = (const float*)d_in[5];
  const float* Wv = (const float*)d_in[6];  const float* bv = (const float*)d_in[7];
  const float* Wo = (const float*)d_in[8];  const float* bo = (const float*)d_in[9];
  const float* W1 = (const float*)d_in[10]; const float* b1 = (const float*)d_in[11];
  const float* W2 = (const float*)d_in[12]; const float* b2 = (const float*)d_in[13];

  char* base = (char*)d_ws;
  size_t off = 0;
  auto alloc = [&](size_t bytes) -> void* {
    void* r = base + off;
    off = (off + bytes + 255) & ~(size_t)255;
    return r;
  };
  u16* hid_bf = (u16*)alloc((size_t)MQ_ * HID_ * 2);
  u16* WqS = (u16*)alloc((size_t)HID_ * HD_ * 2);
  u16* WkS = (u16*)alloc((size_t)HID_ * HD_ * 2);
  u16* WvS = (u16*)alloc((size_t)HID_ * HD_ * 2);
  u16* WoS = (u16*)alloc((size_t)HD_ * HID_ * 2);
  u16* W1S = (u16*)alloc((size_t)HID_ * 1024 * 2);
  u16* W2S = (u16*)alloc((size_t)1024 * HID_ * 2);
  u16* Qp  = (u16*)alloc((size_t)MQ_ * HD_ * 2);
  u16* Kp  = (u16*)alloc((size_t)MQ_ * HD_ * 2);
  u16* Vp  = (u16*)alloc((size_t)MQ_ * HD_ * 2);
  u16* Karea = (u16*)alloc((size_t)B_ * NH_ * MPAD_ * HID_ * 2);
  u16* Varea = (u16*)alloc((size_t)B_ * NH_ * MPAD_ * HID_ * 2);
  u16* ctx   = (u16*)alloc((size_t)MQ_ * HD_ * 2);
  float* attnO  = (float*)alloc((size_t)MQ_ * HID_ * 4);
  float* attn1f = (float*)alloc((size_t)MQ_ * HID_ * 4);
  u16*   attn1b = (u16*)alloc((size_t)MQ_ * HID_ * 2);
  float* attn2f = (float*)alloc((size_t)MQ_ * HID_ * 4);
  u16*   attn2b = (u16*)alloc((size_t)MQ_ * HID_ * 2);
  u16*   h1b    = (u16*)alloc((size_t)MQ_ * 1024 * 2);
  float* h2f    = (float*)alloc((size_t)MQ_ * HID_ * 4);

  const float qscale = 1.0f / 16.0f;  // 1/sqrt(Dh), Dh=256

  f2bf_kernel<<<(MQ_ * HID_ + 255) / 256, 256, 0, stream>>>(hidden, hid_bf, MQ_ * HID_);
  swizzleB_kernel<<<(HID_ * HD_ + 255) / 256, 256, 0, stream>>>(Wq, WqS, HID_, HD_);
  swizzleB_kernel<<<(HID_ * HD_ + 255) / 256, 256, 0, stream>>>(Wk, WkS, HID_, HD_);
  swizzleB_kernel<<<(HID_ * HD_ + 255) / 256, 256, 0, stream>>>(Wv, WvS, HID_, HD_);
  swizzleB_kernel<<<(HD_ * HID_ + 255) / 256, 256, 0, stream>>>(Wo, WoS, HD_, HID_);
  swizzleB_kernel<<<(HID_ * 1024 + 255) / 256, 256, 0, stream>>>(W1, W1S, HID_, 1024);
  swizzleB_kernel<<<(1024 * HID_ + 255) / 256, 256, 0, stream>>>(W2, W2S, 1024, HID_);

  dim3 blk128(128), blk256(256);
  dim3 gQKV(MQ_ / 64, HD_ / 64);
  gemm_bf16_kernel<<<gQKV, blk128, 0, stream>>>(hid_bf, WqS, bq, nullptr, Qp,
                                                MQ_, HD_, HID_, qscale, 0);
  gemm_bf16_kernel<<<gQKV, blk128, 0, stream>>>(hid_bf, WkS, bk, nullptr, Kp,
                                                MQ_, HD_, HID_, 1.f, 0);
  gemm_bf16_kernel<<<gQKV, blk128, 0, stream>>>(hid_bf, WvS, bv, nullptr, Vp,
                                                MQ_, HD_, HID_, 1.f, 0);
  pool_kernel<<<B_ * NH_, blk256, 0, stream>>>(Kp, Vp, Karea, Varea);

  // MHA 1
  flash_kernel<<<dim3(4, B_ * NH_), blk128, 0, stream>>>(Qp, Karea, Varea, ctx);
  gemm_bf16_kernel<<<dim3(MQ_ / 64, HID_ / 64), blk128, 0, stream>>>(
      ctx, WoS, bo, attnO, nullptr, MQ_, HID_, HD_, 1.f, 0);
  addln_kernel<<<MQ_, blk256, 0, stream>>>(attnO, hidden, attn1f, attn1b);

  // MHA 2 (q from attn1; pooled k/v areas reused)
  gemm_bf16_kernel<<<gQKV, blk128, 0, stream>>>(attn1b, WqS, bq, nullptr, Qp,
                                                MQ_, HD_, HID_, qscale, 0);
  flash_kernel<<<dim3(4, B_ * NH_), blk128, 0, stream>>>(Qp, Karea, Varea, ctx);
  gemm_bf16_kernel<<<dim3(MQ_ / 64, HID_ / 64), blk128, 0, stream>>>(
      ctx, WoS, bo, attnO, nullptr, MQ_, HID_, HD_, 1.f, 0);
  addln_kernel<<<MQ_, blk256, 0, stream>>>(attn1f, attnO, attn2f, attn2b);

  // FFN
  gemm_bf16_kernel<<<dim3(MQ_ / 64, 1024 / 64), blk128, 0, stream>>>(
      attn2b, W1S, b1, nullptr, h1b, MQ_, 1024, HID_, 1.f, 1);
  gemm_bf16_kernel<<<dim3(MQ_ / 64, HID_ / 64), blk128, 0, stream>>>(
      h1b, W2S, b2, h2f, nullptr, MQ_, HID_, 1024, 1.f, 0);
  addln_kernel<<<MQ_, blk256, 0, stream>>>(attn2f, h2f, (float*)d_out, nullptr);
}
